// MorphoGPTPredictor_74302934221450
// MI455X (gfx1250) — matrix operand order repaired and checked by reading, not compiled, not run on hardware
//
#include <hip/hip_runtime.h>
#include <math.h>

typedef __attribute__((ext_vector_type(16))) _Float16 v16h;
typedef __attribute__((ext_vector_type(8)))  _Float16 v8h;
typedef __attribute__((ext_vector_type(4)))  _Float16 v4h;
typedef __attribute__((ext_vector_type(8)))  float    v8f;

#define NEG_BIG (-1.0e30f)

// ---------------- constants from the reference ----------------
#define CN 8
#define CS 448
#define CD 768
#define C_VS 50000
#define C_DS 256
#define C_VP 200
#define C_DP 128
#define C_VM 400
#define C_DM 128
#define C_VA 360
#define C_DA 128
#define EPS_LS 0.1f
#define LN_EPS 1e-12f
#define NSPLIT_STEM 16

// =================================================================
// Fragment loaders (CDNA5 WMMA 16x16x32 f16 layouts, ISA 7.12.2)
// =================================================================
// A (16x32, M x K): lane l holds row M=l&15, K = (e&7) + 16*(e>>3) + 8*(l>>4)
// => two contiguous 8-half (16B) chunks at +8*(l>>4) and +16+8*(l>>4).
// B (32x16, K x N): lane l holds col N=l&15, K = (e&7) + 16*(e>>3) + 8*(l>>4)
// => two contiguous 8-half (16B) chunks at +8*(l>>4) and +16+8*(l>>4).
__device__ __forceinline__ v16h load_2x8h(const _Float16* p, int hiOff) {
  v8h lo = *reinterpret_cast<const v8h*>(p);
  v8h hi = *reinterpret_cast<const v8h*>(p + hiOff);
  return __builtin_shufflevector(lo, hi, 0,1,2,3,4,5,6,7,8,9,10,11,12,13,14,15);
}

// ===================== f32 -> f16 bulk convert =====================
__global__ __launch_bounds__(256) void cvt_f16_kernel(
    const float* __restrict__ src, _Float16* __restrict__ dst, int n4) {
  const int i = blockIdx.x * blockDim.x + threadIdx.x;
  if (i >= n4) return;
  const float4 q = *reinterpret_cast<const float4*>(src + (size_t)i * 4);
  v4h o;
  o[0] = (_Float16)q.x; o[1] = (_Float16)q.y;
  o[2] = (_Float16)q.z; o[3] = (_Float16)q.w;
  *reinterpret_cast<v4h*>(dst + (size_t)i * 4) = o;
}

// =================================================================
// 1) Head transform: h = LN(GELU(x @ W + b)) * g + beta  (f16 out)
// =================================================================
template <int DH>
__global__ __launch_bounds__(DH) void head_transform_kernel(
    const float* __restrict__ tr,       // [S, N, D]
    const int*   __restrict__ hid_sel,  // [T]
    const int*   __restrict__ aff_sel,  // [Ta] or null
    int useAff, int Treal,
    const float* __restrict__ W,        // [D, DH]
    const float* __restrict__ bvec,
    const float* __restrict__ gvec,
    const float* __restrict__ betavec,
    _Float16* __restrict__ hbuf)        // [Tpad, DH]
{
  __shared__ float xs[CD];
  __shared__ float red[DH];
  const int t = blockIdx.x;
  const int j = threadIdx.x;
  if (t >= Treal) { hbuf[(size_t)t * DH + j] = (_Float16)0.0f; return; }
  const int i   = useAff ? aff_sel[t] : t;
  const int idx = hid_sel[i];
  const int n = idx / CS;
  const int s = idx - n * CS;
  const float* x = tr + ((size_t)s * CN + n) * CD;
  for (int k = j; k < CD; k += DH) xs[k] = x[k];
  __syncthreads();

  float y = bvec[j];
#pragma unroll 4
  for (int k = 0; k < CD; ++k) y += xs[k] * W[(size_t)k * DH + j];

  const float yg = 0.5f * y * (1.0f + erff(y * 0.70710678118654752f));

  red[j] = yg; __syncthreads();
#pragma unroll
  for (int st = DH / 2; st > 0; st >>= 1) {
    if (j < st) red[j] += red[j + st];
    __syncthreads();
  }
  const float mean = red[0] * (1.0f / DH);
  __syncthreads();
  const float d = yg - mean;
  red[j] = d * d; __syncthreads();
#pragma unroll
  for (int st = DH / 2; st > 0; st >>= 1) {
    if (j < st) red[j] += red[j + st];
    __syncthreads();
  }
  const float var = red[0] * (1.0f / DH);
  const float hv = d * rsqrtf(var + LN_EPS) * gvec[j] + betavec[j];
  hbuf[(size_t)t * DH + j] = (_Float16)hv;
}

// =================================================================
// 2) WMMA decoder + online log-softmax partial stats
//    Block: 256 thr = 8 waves = 4 M-tiles x 2 N-tiles (64 tok x 32 vocab/iter)
//    Branch-free inner loop: vocab index clamped, epilogue predicated.
// =================================================================
template <int DH>
__global__ __launch_bounds__(256) void decoder_stats_kernel(
    const _Float16* __restrict__ hbuf,     // [Tpad, DH]
    const _Float16* __restrict__ emb,      // [V, DH] f16
    const float*    __restrict__ bias,     // [V]
    const int*      __restrict__ tgt_voc,  // [total]
    const int*      __restrict__ tsel,     // [T]
    int T, int Tpad, int V, int chunkV,
    float* __restrict__ stM, float* __restrict__ stS,
    float* __restrict__ stSL, float* __restrict__ stTL)
{
  constexpr int KB = DH / 32;
  __shared__ int   tgt_lds[64];
  __shared__ float redM[2048], redS[2048], redSL[2048], redTL[2048];

  const int tid  = threadIdx.x;
  const int lane = tid & 31, wave = tid >> 5;
  const int Mtile = wave >> 1, Ntile = wave & 1;
  const int tbase = blockIdx.x * 64;
  const int split = blockIdx.y;
  const int vStart = split * chunkV;

  if (tid < 64) {
    const int t = tbase + tid;
    tgt_lds[tid] = (t < T) ? tgt_voc[tsel[t]] : -1;
  }
  __syncthreads();

  const int kh  = lane >> 4;
  const int row = tbase + Mtile * 16 + (lane & 15);
  v16h a[KB];
#pragma unroll
  for (int kb = 0; kb < KB; ++kb)
    a[kb] = load_2x8h(hbuf + (size_t)row * DH + kb * 32 + 8 * kh, 16);

  int myTgt[8];
#pragma unroll
  for (int r = 0; r < 8; ++r) myTgt[r] = tgt_lds[Mtile * 16 + r + 8 * kh];

  float m[8], s[8], sl[8], tl[8];
#pragma unroll
  for (int r = 0; r < 8; ++r) { m[r] = NEG_BIG; s[r] = 0.f; sl[r] = 0.f; tl[r] = NEG_BIG; }

  const int nIt = chunkV / 32;
  for (int it = 0; it < nIt; ++it) {
    const int v = vStart + it * 32 + Ntile * 16 + (lane & 15);
    const bool valid = (v < V);
    const int vc = valid ? v : (V - 1);        // clamped: always safe to load
    const _Float16* eb = emb + (size_t)vc * DH + 8 * kh;
    const float bv = bias[vc];
    v8f c = {};
#pragma unroll
    for (int kb = 0; kb < KB; ++kb) {
      v16h b = load_2x8h(eb + kb * 32, 16);
      c = __builtin_amdgcn_wmma_f32_16x16x32_f16(false, a[kb], false, b,
                                                 (short)0, c, false, false);
    }
    if (valid) {
#pragma unroll
      for (int r = 0; r < 8; ++r) {
        const float l  = c[r] + bv;
        const float mo = m[r];
        const float mn = fmaxf(mo, l);
        s[r]  = s[r] * __expf(mo - mn) + __expf(l - mn);
        m[r]  = mn;
        sl[r] += l;
        if (v == myTgt[r]) tl[r] = l;
      }
    }
  }

  const int base = wave * 256 + lane * 8;
#pragma unroll
  for (int r = 0; r < 8; ++r) {
    redM[base + r] = m[r]; redS[base + r] = s[r];
    redSL[base + r] = sl[r]; redTL[base + r] = tl[r];
  }
  __syncthreads();

  if (tid < 64) {
    const int mt = tid >> 4, mrow = tid & 15;
    const int r2 = mrow & 7, h2 = mrow >> 3;
    float mT = NEG_BIG, tlT = NEG_BIG, slT = 0.f;
    for (int w2 = mt * 2; w2 < mt * 2 + 2; ++w2)
      for (int ln = 0; ln < 16; ++ln) {
        const int ix = w2 * 256 + (h2 * 16 + ln) * 8 + r2;
        mT  = fmaxf(mT, redM[ix]);
        tlT = fmaxf(tlT, redTL[ix]);
        slT += redSL[ix];
      }
    float sT = 0.f;
    for (int w2 = mt * 2; w2 < mt * 2 + 2; ++w2)
      for (int ln = 0; ln < 16; ++ln) {
        const int ix = w2 * 256 + (h2 * 16 + ln) * 8 + r2;
        sT += redS[ix] * __expf(redM[ix] - mT);
      }
    const size_t o = (size_t)split * Tpad + (tbase + tid);
    stM[o] = mT; stS[o] = sT; stSL[o] = slT; stTL[o] = tlT;
  }
}

// ============ combine vocab-split stats -> per-token nll/smooth ============
__global__ __launch_bounds__(256) void combine_kernel(
    const float* __restrict__ stM, const float* __restrict__ stS,
    const float* __restrict__ stSL, const float* __restrict__ stTL,
    int T, int Tpad, int nsplit, int V,
    float* __restrict__ nll, float* __restrict__ sm)
{
  const int t = blockIdx.x * blockDim.x + threadIdx.x;
  if (t >= T) return;
  float mT = NEG_BIG;
  for (int j = 0; j < nsplit; ++j) mT = fmaxf(mT, stM[(size_t)j * Tpad + t]);
  float sT = 0.f, slT = 0.f, tlT = NEG_BIG;
  for (int j = 0; j < nsplit; ++j) {
    const size_t o = (size_t)j * Tpad + t;
    sT  += stS[o] * __expf(stM[o] - mT);
    slT += stSL[o];
    tlT  = fmaxf(tlT, stTL[o]);
  }
  const float lse = mT + logf(sT);
  nll[t] = lse - tlT;
  sm[t]  = (float)V * lse - slT;
}

// ============ WMMA decoder with BCE-with-logits epilogue (affix) ============
template <int DH>
__global__ __launch_bounds__(256) void decoder_bce_kernel(
    const _Float16* __restrict__ hbuf,   // [Tapad, DH]
    const _Float16* __restrict__ emb,    // [VA, DH] f16
    const float*    __restrict__ bias,   // [VA]
    const float*    __restrict__ affp,   // [total, VA]
    const int*      __restrict__ tsel,   // [T]
    const int*      __restrict__ asel,   // [Ta]
    int Ta, int V,
    float* __restrict__ partial)
{
  constexpr int KB = DH / 32;
  __shared__ int   rowOff[64];
  __shared__ float redsum[256];

  const int tid  = threadIdx.x;
  const int lane = tid & 31, wave = tid >> 5;
  const int Mtile = wave >> 1, Ntile = wave & 1;
  const int ibase = blockIdx.x * 64;

  if (tid < 64) {
    const int i = ibase + tid;
    rowOff[tid] = (i < Ta) ? tsel[asel[i]] * V : 0;
  }
  __syncthreads();

  const int kh  = lane >> 4;
  const int row = ibase + Mtile * 16 + (lane & 15);
  v16h a[KB];
#pragma unroll
  for (int kb = 0; kb < KB; ++kb)
    a[kb] = load_2x8h(hbuf + (size_t)row * DH + kb * 32 + 8 * kh, 16);

  int  myOff[8];
  bool myRow[8];
#pragma unroll
  for (int r = 0; r < 8; ++r) {
    const int rib = Mtile * 16 + r + 8 * kh;
    myOff[r] = rowOff[rib];
    myRow[r] = (ibase + rib) < Ta;
  }

  float acc = 0.f;
  const int nIt = (V + 31) / 32;
  for (int it = 0; it < nIt; ++it) {
    const int v = it * 32 + Ntile * 16 + (lane & 15);
    const bool valid = (v < V);
    const int vc = valid ? v : (V - 1);
    const _Float16* eb = emb + (size_t)vc * DH + 8 * kh;
    const float bv = bias[vc];
    v8f c = {};
#pragma unroll
    for (int kb = 0; kb < KB; ++kb) {
      v16h b = load_2x8h(eb + kb * 32, 16);
      c = __builtin_amdgcn_wmma_f32_16x16x32_f16(false, a[kb], false, b,
                                                 (short)0, c, false, false);
    }
    if (valid) {
#pragma unroll
      for (int r = 0; r < 8; ++r) {
        if (myRow[r]) {
          const float z  = c[r] + bv;
          const float ta = affp[(size_t)myOff[r] + v];
          acc += fmaxf(z, 0.f) - z * ta + log1pf(__expf(-fabsf(z)));
        }
      }
    }
  }

  redsum[tid] = acc; __syncthreads();
#pragma unroll
  for (int st = 128; st > 0; st >>= 1) {
    if (tid < st) redsum[tid] += redsum[tid + st];
    __syncthreads();
  }
  if (tid == 0) partial[blockIdx.x] = redsum[0];
}

// ================= final deterministic reduction -> 7 outputs =================
__global__ __launch_bounds__(256) void finalize_kernel(
    const float* __restrict__ nllS, const float* __restrict__ smS,
    const float* __restrict__ nllP, const float* __restrict__ smP,
    const float* __restrict__ nllM, const float* __restrict__ smM,
    const float* __restrict__ bceP, int T, int nBce, float invTaVA,
    float* __restrict__ out)
{
  __shared__ float red[256];
  __shared__ float res[7];
  const int tid = threadIdx.x;
  const float* arrs[6] = {nllS, smS, nllP, smP, nllM, smM};
  for (int k = 0; k < 6; ++k) {
    float a = 0.f;
    for (int i = tid; i < T; i += 256) a += arrs[k][i];
    red[tid] = a; __syncthreads();
#pragma unroll
    for (int st = 128; st > 0; st >>= 1) {
      if (tid < st) red[tid] += red[tid + st];
      __syncthreads();
    }
    if (tid == 0) res[k] = red[0];
    __syncthreads();
  }
  {
    float a = 0.f;
    for (int i = tid; i < nBce; i += 256) a += bceP[i];
    red[tid] = a; __syncthreads();
#pragma unroll
    for (int st = 128; st > 0; st >>= 1) {
      if (tid < st) red[tid] += red[tid + st];
      __syncthreads();
    }
    if (tid == 0) res[6] = red[0];
    __syncthreads();
  }
  if (tid == 0) {
    const float invT = 1.0f / (float)T;
    const float nS = res[0] * invT, sS = res[1] * invT;
    const float nP = res[2] * invT, sP = res[3] * invT;
    const float nM = res[4] * invT, sM = res[5] * invT;
    const float eS = EPS_LS / (float)(C_VS - 1);
    const float eP = EPS_LS / (float)(C_VP - 1);
    const float eM = EPS_LS / (float)(C_VM - 1);
    out[0] = (1.0f - EPS_LS - eS) * nS + eS * sS;
    out[1] = (1.0f - EPS_LS - eP) * nP + eP * sP;
    out[2] = (1.0f - EPS_LS - eM) * nM + eM * sM;
    out[3] = res[6] * invTaVA;
    out[4] = nS;
    out[5] = nP;
    out[6] = nM;
  }
}

// =================================================================
// Host launcher
// =================================================================
extern "C" void kernel_launch(void* const* d_in, const int* in_sizes, int n_in,
                              void* d_out, int out_size, void* d_ws, size_t ws_size,
                              hipStream_t stream) {
  (void)n_in; (void)out_size; (void)ws_size;
  const float* tr    = (const float*)d_in[0];
  const float* affp  = (const float*)d_in[1];
  const int*   stems = (const int*)d_in[2];
  const int*   ptags = (const int*)d_in[3];
  const int*   morfs = (const int*)d_in[4];
  const int*   hid   = (const int*)d_in[5];
  const int*   tsel  = (const int*)d_in[6];
  const int*   asel  = (const int*)d_in[7];
  const float *stem_emb=(const float*)d_in[8],  *stem_W=(const float*)d_in[9],
              *stem_b =(const float*)d_in[10], *stem_g=(const float*)d_in[11],
              *stem_be=(const float*)d_in[12], *stem_bs=(const float*)d_in[13];
  const float *pos_emb=(const float*)d_in[14], *pos_W=(const float*)d_in[15],
              *pos_b  =(const float*)d_in[16], *pos_g=(const float*)d_in[17],
              *pos_be =(const float*)d_in[18], *pos_bs=(const float*)d_in[19];
  const float *mor_emb=(const float*)d_in[20], *mor_W=(const float*)d_in[21],
              *mor_b  =(const float*)d_in[22], *mor_g=(const float*)d_in[23],
              *mor_be =(const float*)d_in[24], *mor_bs=(const float*)d_in[25];
  const float *aff_emb=(const float*)d_in[26], *aff_W=(const float*)d_in[27],
              *aff_b  =(const float*)d_in[28], *aff_g=(const float*)d_in[29],
              *aff_be =(const float*)d_in[30], *aff_bs=(const float*)d_in[31];

  const int T  = in_sizes[5];
  const int Ta = in_sizes[7];
  const int Tpad  = ((T  + 63) / 64) * 64;
  const int Tapad = ((Ta + 63) / 64) * 64 > 0 ? ((Ta + 63) / 64) * 64 : 64;
  const int nBce  = Tapad / 64;
  const int chunkS = (((C_VS + NSPLIT_STEM - 1) / NSPLIT_STEM) + 31) / 32 * 32;
  const int chunkP = ((C_VP + 31) / 32) * 32;
  const int chunkM = ((C_VM + 31) / 32) * 32;

  char* base = (char*)d_ws;
  size_t off = 0;
  auto alloc = [&](size_t bytes) -> void* {
    void* p = base + off;
    off += (bytes + 255) & ~(size_t)255;
    return p;
  };
  _Float16* hbS = (_Float16*)alloc((size_t)Tpad  * C_DS * 2);
  _Float16* hbP = (_Float16*)alloc((size_t)Tpad  * C_DP * 2);
  _Float16* hbM = (_Float16*)alloc((size_t)Tpad  * C_DM * 2);
  _Float16* hbA = (_Float16*)alloc((size_t)Tapad * C_DA * 2);
  _Float16* embS16 = (_Float16*)alloc((size_t)C_VS * C_DS * 2);
  _Float16* embP16 = (_Float16*)alloc((size_t)C_VP * C_DP * 2);
  _Float16* embM16 = (_Float16*)alloc((size_t)C_VM * C_DM * 2);
  _Float16* embA16 = (_Float16*)alloc((size_t)C_VA * C_DA * 2);
  float* sM_S  = (float*)alloc((size_t)NSPLIT_STEM * Tpad * 4);
  float* sS_S  = (float*)alloc((size_t)NSPLIT_STEM * Tpad * 4);
  float* sSL_S = (float*)alloc((size_t)NSPLIT_STEM * Tpad * 4);
  float* sTL_S = (float*)alloc((size_t)NSPLIT_STEM * Tpad * 4);
  float* sM_P  = (float*)alloc((size_t)Tpad * 4);
  float* sS_P  = (float*)alloc((size_t)Tpad * 4);
  float* sSL_P = (float*)alloc((size_t)Tpad * 4);
  float* sTL_P = (float*)alloc((size_t)Tpad * 4);
  float* sM_M  = (float*)alloc((size_t)Tpad * 4);
  float* sS_M  = (float*)alloc((size_t)Tpad * 4);
  float* sSL_M = (float*)alloc((size_t)Tpad * 4);
  float* sTL_M = (float*)alloc((size_t)Tpad * 4);
  float* nllS = (float*)alloc((size_t)Tpad * 4);
  float* smS  = (float*)alloc((size_t)Tpad * 4);
  float* nllP = (float*)alloc((size_t)Tpad * 4);
  float* smP  = (float*)alloc((size_t)Tpad * 4);
  float* nllM = (float*)alloc((size_t)Tpad * 4);
  float* smM  = (float*)alloc((size_t)Tpad * 4);
  float* bceP = (float*)alloc((size_t)nBce * 4);

  // 0) embedding tables f32 -> f16 (stem table lives in L2 thereafter)
  {
    const int nS4 = C_VS * C_DS / 4, nP4 = C_VP * C_DP / 4,
              nM4 = C_VM * C_DM / 4, nA4 = C_VA * C_DA / 4;
    cvt_f16_kernel<<<(nS4 + 255) / 256, 256, 0, stream>>>(stem_emb, embS16, nS4);
    cvt_f16_kernel<<<(nP4 + 255) / 256, 256, 0, stream>>>(pos_emb,  embP16, nP4);
    cvt_f16_kernel<<<(nM4 + 255) / 256, 256, 0, stream>>>(mor_emb,  embM16, nM4);
    cvt_f16_kernel<<<(nA4 + 255) / 256, 256, 0, stream>>>(aff_emb,  embA16, nA4);
  }

  // 1) head transforms
  head_transform_kernel<C_DS><<<Tpad, C_DS, 0, stream>>>(
      tr, hid, nullptr, 0, T, stem_W, stem_b, stem_g, stem_be, hbS);
  head_transform_kernel<C_DP><<<Tpad, C_DP, 0, stream>>>(
      tr, hid, nullptr, 0, T, pos_W, pos_b, pos_g, pos_be, hbP);
  head_transform_kernel<C_DM><<<Tpad, C_DM, 0, stream>>>(
      tr, hid, nullptr, 0, T, mor_W, mor_b, mor_g, mor_be, hbM);
  head_transform_kernel<C_DA><<<Tapad, C_DA, 0, stream>>>(
      tr, hid, asel, 1, Ta, aff_W, aff_b, aff_g, aff_be, hbA);

  // 2) WMMA decoders + softmax stats
  decoder_stats_kernel<C_DS><<<dim3(Tpad / 64, NSPLIT_STEM), 256, 0, stream>>>(
      hbS, embS16, stem_bs, stems, tsel, T, Tpad, C_VS, chunkS,
      sM_S, sS_S, sSL_S, sTL_S);
  decoder_stats_kernel<C_DP><<<dim3(Tpad / 64, 1), 256, 0, stream>>>(
      hbP, embP16, pos_bs, ptags, tsel, T, Tpad, C_VP, chunkP,
      sM_P, sS_P, sSL_P, sTL_P);
  decoder_stats_kernel<C_DM><<<dim3(Tpad / 64, 1), 256, 0, stream>>>(
      hbM, embM16, mor_bs, morfs, tsel, T, Tpad, C_VM, chunkM,
      sM_M, sS_M, sSL_M, sTL_M);

  // 3) combine splits
  const int cgrid = (T + 255) / 256;
  combine_kernel<<<cgrid, 256, 0, stream>>>(sM_S, sS_S, sSL_S, sTL_S,
                                            T, Tpad, NSPLIT_STEM, C_VS, nllS, smS);
  combine_kernel<<<cgrid, 256, 0, stream>>>(sM_P, sS_P, sSL_P, sTL_P,
                                            T, Tpad, 1, C_VP, nllP, smP);
  combine_kernel<<<cgrid, 256, 0, stream>>>(sM_M, sS_M, sSL_M, sTL_M,
                                            T, Tpad, 1, C_VM, nllM, smM);

  // 4) affix BCE
  decoder_bce_kernel<C_DA><<<nBce, 256, 0, stream>>>(
      hbA, embA16, aff_bs, affp, tsel, asel, Ta, C_VA, bceP);

  // 5) finalize
  const float invTaVA = (Ta > 0) ? 1.0f / ((float)Ta * (float)C_VA) : 0.0f;
  finalize_kernel<<<1, 256, 0, stream>>>(nllS, smS, nllP, smP, nllM, smM,
                                         bceP, T, nBce, invTaVA, (float*)d_out);
}
